// MetaKernel_29618094473257
// MI455X (gfx1250) — matrix operand order repaired and checked by reading, not compiled, not run on hardware
//
#include <hip/hip_runtime.h>
#include <hip/hip_bf16.h>

typedef __bf16 v16bf __attribute__((ext_vector_type(16)));
typedef __bf16 v8bf  __attribute__((ext_vector_type(8)));
typedef float  v8f   __attribute__((ext_vector_type(8)));
typedef int    v4i_vs __attribute__((vector_size(16)));
typedef v4i_vs __attribute__((address_space(1))) gv4i;   // global int4
typedef v4i_vs __attribute__((address_space(3))) lv4i;   // LDS int4

#define Bn    2
#define Hh    64
#define Ww    2048
#define CIN   64
#define COUT  64
#define CTOT  68
#define Ff    576          // 9 * CIN
#define TP    64           // positions per workgroup
#define ASTR  584          // padded LDS row stride (bf16 elems): 292 dwords == 36 mod 64 -> conflict-free b128
#define LSTR  13
#define EPSV  1e-5f

// ---- async global->LDS B128 (ASYNCcnt path, gfx1250) ----
static __device__ __forceinline__ void async_g2l_b128(const void* gsrc, void* ldst)
{
    unsigned lds_off = (unsigned)(size_t)ldst;   // low 32 bits of flat shared addr == LDS offset
#if __has_builtin(__builtin_amdgcn_global_load_async_to_lds_b128)
    __builtin_amdgcn_global_load_async_to_lds_b128(
        (gv4i*)(size_t)gsrc,
        (lv4i*)(size_t)lds_off,
        0, 0);
#else
    asm volatile("global_load_async_to_lds_b128 %0, %1, off"
                 :: "v"(lds_off), "v"((unsigned long long)(size_t)gsrc)
                 : "memory");
#endif
}

static __device__ __forceinline__ void wait_asynccnt0()
{
#if __has_builtin(__builtin_amdgcn_s_wait_asynccnt)
    __builtin_amdgcn_s_wait_asynccnt(0);
#else
    asm volatile("s_wait_asynccnt 0x0" ::: "memory");
#endif
}

// ---------------- prepass: fold BN params, convert Wagg to bf16 ----------------
__global__ void prep_kernel(const float* __restrict__ Wagg,
                            const float* __restrict__ W1,
                            const float* __restrict__ g1, const float* __restrict__ b1,
                            const float* __restrict__ m1, const float* __restrict__ v1,
                            const float* __restrict__ g2, const float* __restrict__ b2n,
                            const float* __restrict__ m2, const float* __restrict__ v2,
                            __bf16* __restrict__ waggbf, float* __restrict__ w1f,
                            float* __restrict__ b1f, float* __restrict__ s2,
                            float* __restrict__ t2)
{
    int idx = blockIdx.x * 256 + threadIdx.x;
    if (idx < COUT * Ff) waggbf[idx] = (__bf16)Wagg[idx];
    if (idx < CIN) {
        float a = g1[idx] * rsqrtf(v1[idx] + EPSV);
#pragma unroll
        for (int m = 0; m < 4; ++m) w1f[m * CIN + idx] = W1[idx * 4 + m] * a;
        b1f[idx] = b1[idx] - m1[idx] * a;
        float s = g2[idx] * rsqrtf(v2[idx] + EPSV);
        s2[idx] = s;
        t2[idx] = b2n[idx] - m2[idx] * s;
    }
}

// ---------------- main fused kernel ----------------
__global__ __launch_bounds__(256)
void paconv_kernel(const float* __restrict__ x, const int* __restrict__ mask,
                   const float* __restrict__ W2, const float* __restrict__ b2,
                   const __bf16* __restrict__ waggbf,
                   const float* __restrict__ w1fg, const float* __restrict__ b1fg,
                   const float* __restrict__ s2g,  const float* __restrict__ t2g,
                   float* __restrict__ out)
{
    __shared__ __align__(16) __bf16 ldsA[TP * ASTR];     // A tile (bf16), reused as f32 out staging
    __shared__ __align__(16) __bf16 ldsW[COUT * ASTR];   // Wagg bf16, padded rows (async DMA target)
    __shared__ __align__(16) float  ldsLg[TP * LSTR];    // logits -> softmax weights
    __shared__ __align__(16) float  ldsPnc[TP * 5];      // center positional feats
    __shared__ __align__(16) float  ldsW1f[4 * CIN];     // folded W1 (m-major)
    __shared__ __align__(16) float  ldsB1f[CIN];
    __shared__ __align__(16) float  ldsW2[CIN];

    const int tid = threadIdx.x;
    const int wg  = blockIdx.x;
    const int tw  = wg & 31;
    const int h   = (wg >> 5) & 63;
    const int b   = wg >> 11;
    const int w0  = tw << 6;

    // ---- S0: kick off async DMA of Wagg (bf16) into padded LDS rows.
    // 4608 x 16B transfers; lands in LDS while the attention phase runs.
    for (int idx = tid; idx < COUT * 72; idx += 256) {
        int o = idx / 72, j = idx - o * 72;
        async_g2l_b128(waggbf + o * Ff + j * 8, ldsW + o * ASTR + j * 8);
    }
    // folded-param copies (small, synchronous)
    ldsW1f[tid] = w1fg[tid];
    if (tid < CIN) { ldsB1f[tid] = b1fg[tid]; ldsW2[tid] = W2[tid]; }
    // ---- S1: center positional features (coalesced: 64 lanes along w) ----
    {
        int m = tid >> 6, n = tid & 63;
        ldsPnc[n * 5 + m] = x[(((size_t)b * CTOT + m) * Hh + h) * Ww + w0 + n];
    }
    __syncthreads();

    // ---- S3: masked attention logits: (n,k) pairs over all 256 threads ----
    const float b2v = b2[0];
    for (int p = tid; p < TP * 9; p += 256) {
        int n   = p & 63, k = p >> 6;
        int hh  = h + k / 3 - 1;
        int ww2 = w0 + n + (k % 3) - 1;
        float logit = 0.f;
        if (hh >= 0 && hh < Hh && ww2 >= 0 && ww2 < Ww) {
            int mk = mask[((size_t)b * Hh + hh) * Ww + ww2];
            if (mk != 0) {
                size_t base = (((size_t)b * CTOT) * Hh + hh) * Ww + ww2;
                float d0 = x[base]                       - ldsPnc[n * 5 + 0];
                float d1 = x[base + (size_t)Hh * Ww]     - ldsPnc[n * 5 + 1];
                float d2 = x[base + 2 * (size_t)Hh * Ww] - ldsPnc[n * 5 + 2];
                float d3 = x[base + 3 * (size_t)Hh * Ww] - ldsPnc[n * 5 + 3];
                float acc = b2v;
#pragma unroll 8
                for (int c2 = 0; c2 < 32; ++c2) {
                    float2 a0 = ((const float2*)(ldsW1f      ))[c2];
                    float2 a1 = ((const float2*)(ldsW1f +  64))[c2];
                    float2 a2 = ((const float2*)(ldsW1f + 128))[c2];
                    float2 a3 = ((const float2*)(ldsW1f + 192))[c2];
                    float2 bb = ((const float2*)ldsB1f)[c2];
                    float hx = fmaf(d3, a3.x, fmaf(d2, a2.x, fmaf(d1, a1.x, fmaf(d0, a0.x, bb.x))));
                    float hy = fmaf(d3, a3.y, fmaf(d2, a2.y, fmaf(d1, a1.y, fmaf(d0, a0.y, bb.y))));
                    float2 wv = ((const float2*)ldsW2)[c2];
                    acc = fmaf(wv.x, fmaxf(hx, 0.f), acc);
                    acc = fmaf(wv.y, fmaxf(hy, 0.f), acc);
                }
                logit = acc;
            }
        }
        ldsLg[n * LSTR + k] = logit;   // OOB / masked taps contribute exp(0) like the reference
    }
    __syncthreads();

    // ---- S5: softmax over the 9 taps ----
    if (tid < TP) {
        float v[9], mx = -3.4e38f;
#pragma unroll
        for (int k = 0; k < 9; ++k) { v[k] = ldsLg[tid * LSTR + k]; mx = fmaxf(mx, v[k]); }
        float s = 0.f;
#pragma unroll
        for (int k = 0; k < 9; ++k) { v[k] = __expf(v[k] - mx); s += v[k]; }
        float inv = 1.f / s;
#pragma unroll
        for (int k = 0; k < 9; ++k) ldsLg[tid * LSTR + k] = v[k] * inv;
    }
    __syncthreads();

    // ---- S7: build A[n][k*64+c] = w_k * feat_kc  (bf16) ----
    {
        int grp = tid >> 6, n = tid & 63;
        for (int j = grp; j < 9 * CIN; j += 4) {
            int k = j >> 6, c = j & 63;
            int hh  = h + k / 3 - 1;
            int ww2 = w0 + n + (k % 3) - 1;
            float f = 0.f;
            if (hh >= 0 && hh < Hh && ww2 >= 0 && ww2 < Ww)
                f = x[(((size_t)b * CTOT + 4 + c) * Hh + hh) * Ww + ww2];
            float wk = ldsLg[n * LSTR + k];
            ldsA[n * ASTR + j] = (__bf16)(wk * f);
        }
    }
    wait_asynccnt0();      // Wagg DMA must have landed in ldsW before anyone reads it
    __syncthreads();

    // ---- S9: WMMA  y[64 pos x 64 ch] = A[64x576] x Wagg^T[576x64]  (bf16 -> f32) ----
    const int wv   = tid >> 5;
    const int lane = tid & 31;
    const int l    = lane & 15;
    const int hi   = lane >> 4;
    const int mt   = wv & 3;          // M-tile (16 positions)
    const int ng   = wv >> 2;         // N-tile pair
    const int o0   = ng * 32 + l;     // output channel for acc0 (per C/D layout, N = lane%16)
    const int o1   = o0 + 16;

    const __bf16* arow  = ldsA + (mt * 16 + l) * ASTR;
    const __bf16* brow0 = ldsW + o0 * ASTR;
    const __bf16* brow1 = ldsW + o1 * ASTR;

    v8f acc0 = {}, acc1 = {};
#pragma unroll 3
    for (int kc = 0; kc < 18; ++kc) {
        int f0 = kc * 32;
        // A frag: lanes 0-15 hold K 0..15, lanes 16-31 hold K 16..31
        v8bf alo = *(const v8bf*)(arow + f0 + hi * 16);
        v8bf ahi = *(const v8bf*)(arow + f0 + hi * 16 + 8);
        v16bf av;
#pragma unroll
        for (int i = 0; i < 8; ++i) { av[i] = alo[i]; av[i + 8] = ahi[i]; }
        // B frag: lanes 0-15 hold K 0..15, lanes 16-31 hold K 16..31 (column = lane%16)
        v8bf b0lo = *(const v8bf*)(brow0 + f0 + hi * 16);
        v8bf b0hi = *(const v8bf*)(brow0 + f0 + hi * 16 + 8);
        v8bf b1lo = *(const v8bf*)(brow1 + f0 + hi * 16);
        v8bf b1hi = *(const v8bf*)(brow1 + f0 + hi * 16 + 8);
        v16bf bv0, bv1;
#pragma unroll
        for (int i = 0; i < 8; ++i) {
            bv0[i] = b0lo[i]; bv0[i + 8] = b0hi[i];
            bv1[i] = b1lo[i]; bv1[i + 8] = b1hi[i];
        }
        acc0 = __builtin_amdgcn_wmma_f32_16x16x32_bf16(false, av, false, bv0, (short)0, acc0, false, false);
        acc1 = __builtin_amdgcn_wmma_f32_16x16x32_bf16(false, av, false, bv1, (short)0, acc1, false, false);
    }

    // ---- S10: BN2 + ReLU, transpose via LDS, coalesced global store ----
    float sa = s2g[o0], ta = t2g[o0];
    float sb = s2g[o1], tb = t2g[o1];
    __syncthreads();                       // all WMMA reads of ldsA complete
    float* outb = (float*)ldsA;            // [COUT][68] staging (17408 B <= A region)
#pragma unroll
    for (int r = 0; r < 8; ++r) {
        int pos = mt * 16 + hi * 8 + r;    // C/D layout: VGPR r -> M = r + 8*hi
        outb[o0 * 68 + pos] = fmaxf(fmaf(acc0[r], sa, ta), 0.f);
        outb[o1 * 68 + pos] = fmaxf(fmaf(acc1[r], sb, tb), 0.f);
    }
    __syncthreads();
    for (int idx = tid; idx < COUT * TP; idx += 256) {
        int o = idx >> 6, n = idx & 63;
        out[(((size_t)b * COUT + o) * Hh + h) * Ww + w0 + n] = outb[o * 68 + n];
    }
}

extern "C" void kernel_launch(void* const* d_in, const int* in_sizes, int n_in,
                              void* d_out, int out_size, void* d_ws, size_t ws_size,
                              hipStream_t stream)
{
    const float* x    = (const float*)d_in[0];
    const int*   mask = (const int*)d_in[1];
    const float* W1   = (const float*)d_in[2];
    const float* g1   = (const float*)d_in[3];
    const float* b1   = (const float*)d_in[4];
    const float* m1   = (const float*)d_in[5];
    const float* v1   = (const float*)d_in[6];
    const float* W2   = (const float*)d_in[7];
    const float* b2   = (const float*)d_in[8];
    const float* Wagg = (const float*)d_in[9];
    const float* g2   = (const float*)d_in[10];
    const float* b2n  = (const float*)d_in[11];
    const float* m2   = (const float*)d_in[12];
    const float* v2   = (const float*)d_in[13];
    float* out = (float*)d_out;

    // workspace carve: [0, 73728) Wagg bf16; then folded f32 params
    __bf16* waggbf = (__bf16*)d_ws;
    float*  fws    = (float*)((char*)d_ws + COUT * Ff * 2);
    float*  w1f = fws;           // 256 floats
    float*  b1f = fws + 256;     // 64
    float*  s2  = fws + 320;     // 64
    float*  t2  = fws + 384;     // 64

    prep_kernel<<<(COUT * Ff + 255) / 256, 256, 0, stream>>>(
        Wagg, W1, g1, b1, m1, v1, g2, b2n, m2, v2, waggbf, w1f, b1f, s2, t2);

    paconv_kernel<<<Bn * Hh * (Ww / TP), 256, 0, stream>>>(
        x, mask, W2, b2, waggbf, w1f, b1f, s2, t2, out);
}
